// DecisionTransformer_47863115547319
// MI455X (gfx1250) — compile-verified, matrix operand order checked by reading, not run on hardware
//
#include <hip/hip_runtime.h>
#include <stdint.h>

// ---------------- model dims ----------------
#define H_    768
#define T_    30
#define L_    12
#define NH_   12
#define D_    64
#define T3_   90
#define B_    128
#define ROWS_ (B_ * T3_)     // 11520
#define H3_   2304
#define H4_   3072
#define SD2_  7056
#define FLAT_ 3136
#define HT_   23040          // H*T

// ---------------- WMMA types ----------------
typedef __attribute__((ext_vector_type(16))) __bf16 v16bf;
typedef __attribute__((ext_vector_type(8)))  float  v8f;

static __device__ __forceinline__ uint16_t f2bfu(float f) {
  union { float f; uint32_t u; } v; v.f = f;
  uint32_t r = v.u + 0x7FFFu + ((v.u >> 16) & 1u);   // round-to-nearest-even
  return (uint16_t)(r >> 16);
}
static __device__ __forceinline__ float bfu2f(uint16_t s) {
  union { uint32_t u; float f; } v; v.u = ((uint32_t)s) << 16; return v.f;
}
static __device__ __forceinline__ v8f vzero8() {
  v8f z = {0.f, 0.f, 0.f, 0.f, 0.f, 0.f, 0.f, 0.f};
  return z;
}
// ISA 16-bit A-fragment load: rowBase points at the 32 contiguous halves of this
// lane's matrix row (K-slice). lanes 0-15: K{0..7,16..23}; lanes 16-31: K{8..15,24..31}.
static __device__ __forceinline__ v16bf frag_ld(const uint16_t* rowBase, int laneHi) {
  const uint32_t* p = (const uint32_t*)rowBase + (laneHi ? 4 : 0);
  union { uint32_t u[8]; v16bf v; } f;
#pragma unroll
  for (int i = 0; i < 4; ++i) { f.u[i] = p[i]; f.u[4 + i] = p[8 + i]; }
  return f.v;
}
static __device__ __forceinline__ v8f wmma_bf(v16bf a, v16bf b, v8f c) {
  return __builtin_amdgcn_wmma_f32_16x16x32_bf16(false, a, false, b, (short)0, c,
                                                 false, false);
}

// ---------------- gfx1250 async memory->LDS path (guarded) ----------------
#if __has_builtin(__builtin_amdgcn_global_load_async_to_lds_b128)
#define HAVE_ASYNC_LDS 1
typedef int v4i_ __attribute__((vector_size(16)));   // b128 payload type
static __device__ __forceinline__ void async_g2l_b128(const void* g, void* l) {
  __builtin_amdgcn_global_load_async_to_lds_b128(
      (__attribute__((address_space(1))) v4i_*)(g),
      (__attribute__((address_space(3))) v4i_*)(l), 0, 0);
}
static __device__ __forceinline__ void wait_async0() {
#if __has_builtin(__builtin_amdgcn_s_wait_asynccnt)
  __builtin_amdgcn_s_wait_asynccnt(0);
#else
  asm volatile("s_wait_asynccnt 0x0" ::: "memory");
#endif
}
#else
#define HAVE_ASYNC_LDS 0
#endif

// ============================================================
// Generic bf16 WMMA GEMM:  out = act(A[MxK]bf16 @ W[KxN]f32 + bias + resid)
// BM=128 BN=128 BK=64, 256 threads = 8 waves (4 along M x 2 along N),
// each wave computes 32x64 = 2x4 WMMA tiles, 16 WMMAs per stage.
// Requires M % 128 == 0, K % 64 == 0; N is guarded.
// act: 0=none, 1=tanh, 2=gelu(exact)
// ============================================================
__global__ __launch_bounds__(256)
void gemm_wmma(const uint16_t* __restrict__ A, const float* __restrict__ W,
               const float* __restrict__ bias, const float* __restrict__ resid,
               float* __restrict__ outF, uint16_t* __restrict__ outB,
               int M, int N, int K, int act)
{
  __shared__ __align__(16) uint16_t As[128 * 64];   // [m][k]  16 KB
  __shared__ __align__(16) uint16_t Ws[128 * 64];   // [n][k]  16 KB (transposed)
  const int tid  = threadIdx.x;
  const int lane = tid & 31, wid = tid >> 5;
  const int wm = wid >> 1, wn = wid & 1;
  const int l15 = lane & 15, hi = (lane >> 4) & 1;
  const int row0 = blockIdx.y * 128;
  const int col0 = blockIdx.x * 128;

  v8f acc[2][4];
#pragma unroll
  for (int i = 0; i < 2; ++i)
#pragma unroll
    for (int j = 0; j < 4; ++j) acc[i][j] = vzero8();

  for (int k0 = 0; k0 < K; k0 += 64) {
    // ---- stage A tile 128x64 bf16: 1024 x 16B chunks ----
#if HAVE_ASYNC_LDS
#pragma unroll
    for (int i = 0; i < 4; ++i) {
      int e = i * 256 + tid;
      int r = e >> 3, seg = e & 7;
      async_g2l_b128(A + (size_t)(row0 + r) * K + k0 + seg * 8,
                     As + r * 64 + seg * 8);
    }
#else
#pragma unroll
    for (int i = 0; i < 4; ++i) {
      int e = i * 256 + tid;
      int r = e >> 3, seg = e & 7;
      const uint32_t* src = (const uint32_t*)(A + (size_t)(row0 + r) * K + k0) + seg * 4;
      uint32_t* dst = (uint32_t*)(As + r * 64) + seg * 4;
      dst[0] = src[0]; dst[1] = src[1]; dst[2] = src[2]; dst[3] = src[3];
    }
#endif
    // ---- stage W tile 64x128 f32 -> bf16 transposed ----
#pragma unroll
    for (int i = 0; i < 8; ++i) {
      int e = i * 256 + tid;                 // 2048 float4 quads
      int kk = e >> 5;                       // 64 k-rows, 32 quads each
      int n4 = (e & 31) * 4;
      int gn = col0 + n4;
      const float* wp = W + (size_t)(k0 + kk) * N;
      float w0 = 0.f, w1 = 0.f, w2 = 0.f, w3 = 0.f;
      if (gn + 3 < N) {
        const float4 q = *(const float4*)(wp + gn);
        w0 = q.x; w1 = q.y; w2 = q.z; w3 = q.w;
      } else {
        if (gn     < N) w0 = wp[gn];
        if (gn + 1 < N) w1 = wp[gn + 1];
        if (gn + 2 < N) w2 = wp[gn + 2];
        if (gn + 3 < N) w3 = wp[gn + 3];
      }
      uint16_t* c = Ws + n4 * 64 + kk;
      c[0] = f2bfu(w0); c[64] = f2bfu(w1); c[128] = f2bfu(w2); c[192] = f2bfu(w3);
    }
    if (k0 + 64 < K)   // gfx1250 global_prefetch_b8 for next W tile
      __builtin_prefetch(W + (size_t)(k0 + 64) * N + col0 + (tid & 127), 0, 0);
#if HAVE_ASYNC_LDS
    wait_async0();
#endif
    __syncthreads();

    const uint16_t* arow = As + (wm * 32 + l15) * 64;
#pragma unroll
    for (int kk = 0; kk < 64; kk += 32) {
      v16bf a0 = frag_ld(arow + kk, hi);
      v16bf a1 = frag_ld(arow + 16 * 64 + kk, hi);
#pragma unroll
      for (int nt = 0; nt < 4; ++nt) {
        v16bf bf = frag_ld(Ws + (wn * 64 + nt * 16 + l15) * 64 + kk, hi);
        acc[0][nt] = wmma_bf(a0, bf, acc[0][nt]);
        acc[1][nt] = wmma_bf(a1, bf, acc[1][nt]);
      }
    }
    __syncthreads();
  }

  // epilogue: C layout = lane -> column, vgpr r -> row (+8 for hi lanes)
#pragma unroll
  for (int mt = 0; mt < 2; ++mt) {
#pragma unroll
    for (int nt = 0; nt < 4; ++nt) {
      int col = col0 + wn * 64 + nt * 16 + l15;
      if (col >= N) continue;
      float bv = bias ? bias[col] : 0.0f;
#pragma unroll
      for (int r = 0; r < 8; ++r) {
        int row = row0 + wm * 32 + mt * 16 + hi * 8 + r;
        float v = acc[mt][nt][r] + bv;
        if (resid) v += resid[(size_t)row * N + col];
        if (act == 1) v = tanhf(v);
        else if (act == 2) v = 0.5f * v * (1.0f + erff(v * 0.70710678f));
        if (outF) outF[(size_t)row * N + col] = v;
        if (outB) outB[(size_t)row * N + col] = f2bfu(v);
      }
    }
  }
}

// ============================================================
// Fused causal attention per (batch, head). qkv bf16 (ROWS x 2304),
// y bf16 (ROWS x 768). One 256-thread block per (b,h); WMMA for QK^T and P.V.
// T3=90 padded to 96; D=64; scale=1/8.
// ============================================================
__global__ __launch_bounds__(256)
void attn_wmma(const uint16_t* __restrict__ qkv, uint16_t* __restrict__ y)
{
  __shared__ __align__(16) uint16_t sm[30720];   // 60 KB
  uint16_t* Qs = sm;                             // 96x64 bf16 (phase 1)
  uint16_t* Ks = sm + 96 * 64;                   // 96x64 bf16 (phase 1)
  float*    Ss = (float*)(sm + 2 * 96 * 64);     // 96x96 f32
  uint16_t* Ps = sm;                             // 96x96 bf16 (phase 2, over Q/K)
  uint16_t* Vs = sm + 96 * 96;                   // 64x96 bf16 d-major (phase 2)

  const int tid  = threadIdx.x;
  const int lane = tid & 31, wid = tid >> 5;
  const int l15 = lane & 15, hi = (lane >> 4) & 1;
  const int b = blockIdx.x / NH_;
  const int h = blockIdx.x % NH_;

  // ---- load Q and K tiles (rows >= 90 zero-padded) ----
#if HAVE_ASYNC_LDS
  if (tid < 192) {                               // zero-pad rows 90..95
    ((uint32_t*)Qs)[2880 + tid] = 0;
    ((uint32_t*)Ks)[2880 + tid] = 0;
  }
  for (int e = tid; e < 720; e += 256) {         // 90 rows x 8 chunks of 16B
    int r = e >> 3, seg = e & 7;
    const uint16_t* gq = qkv + (size_t)(b * T3_ + r) * H3_ + h * D_ + seg * 8;
    async_g2l_b128(gq,       Qs + r * 64 + seg * 8);
    async_g2l_b128(gq + 768, Ks + r * 64 + seg * 8);
  }
  wait_async0();
#else
#pragma unroll
  for (int i = 0; i < 12; ++i) {
    int d = i * 256 + tid;                       // 3072 dwords each
    int r = d >> 5, c = d & 31;
    uint32_t qv = 0, kv = 0;
    if (r < T3_) {
      const uint32_t* p =
          (const uint32_t*)(qkv + (size_t)(b * T3_ + r) * H3_ + h * D_);
      qv = p[c];
      kv = p[c + 384];                           // +768 halves = K block
    }
    ((uint32_t*)Qs)[d] = qv;
    ((uint32_t*)Ks)[d] = kv;
  }
#endif
  __syncthreads();

  // S = scale * Q K^T, causal mask. 36 tiles of 16x16, 2 K-steps of 32.
  for (int t = wid; t < 36; t += 8) {
    int mi = t / 6, ni = t % 6;
    v8f acc = vzero8();
#pragma unroll
    for (int kk = 0; kk < 64; kk += 32) {
      v16bf a  = frag_ld(Qs + (mi * 16 + l15) * 64 + kk, hi);
      v16bf bb = frag_ld(Ks + (ni * 16 + l15) * 64 + kk, hi);
      acc = wmma_bf(a, bb, acc);
    }
#pragma unroll
    for (int r = 0; r < 8; ++r) {
      int q  = mi * 16 + hi * 8 + r;
      int kc = ni * 16 + l15;
      float sv = acc[r] * 0.125f;
      Ss[q * 96 + kc] = (kc <= q && kc < T3_) ? sv : -1e30f;
    }
  }
  __syncthreads();

  // row softmax -> P bf16
  if (tid < 96) {
    float* srow = Ss + tid * 96;
    float m = -1e30f;
    for (int c = 0; c < 96; ++c) m = fmaxf(m, srow[c]);
    float sum = 0.0f;
    for (int c = 0; c < 96; ++c) sum += __expf(srow[c] - m);
    float inv = 1.0f / sum;
    uint16_t* prow = Ps + tid * 96;
    for (int c = 0; c < 96; ++c) prow[c] = f2bfu(__expf(srow[c] - m) * inv);
  }
  __syncthreads();

  // stage V transposed (d-major) so P.V B-fragment is contiguous
#pragma unroll
  for (int i = 0; i < 24; ++i) {
    int e = i * 256 + tid;                       // 6144 halves
    int dd = e / 96, key = e % 96;
    uint16_t vv = 0;
    if (key < T3_)
      vv = qkv[(size_t)(b * T3_ + key) * H3_ + 1536 + h * D_ + dd];
    Vs[dd * 96 + key] = vv;
  }
  __syncthreads();

  // O = P @ V : 6x4 tiles, 3 K-steps of 32
  for (int t = wid; t < 24; t += 8) {
    int mi = t >> 2, ni = t & 3;
    v8f acc = vzero8();
#pragma unroll
    for (int kk = 0; kk < 96; kk += 32) {
      v16bf a  = frag_ld(Ps + (mi * 16 + l15) * 96 + kk, hi);
      v16bf bb = frag_ld(Vs + (ni * 16 + l15) * 96 + kk, hi);
      acc = wmma_bf(a, bb, acc);
    }
#pragma unroll
    for (int r = 0; r < 8; ++r) {
      int q = mi * 16 + hi * 8 + r;
      if (q < T3_) {
        int dcol = ni * 16 + l15;
        y[(size_t)(b * T3_ + q) * H_ + h * D_ + dcol] = f2bfu(acc[r]);
      }
    }
  }
}

// ============================================================
// Conv encoder (direct VALU, small FLOP fraction)
// ============================================================
__global__ __launch_bounds__(256)
void conv1_k(const float* __restrict__ in, const float* __restrict__ w,
             const float* __restrict__ b, float* __restrict__ out)
{
  int i = blockIdx.x * 256 + threadIdx.x;        // B*32*20*20
  if (i >= B_ * 32 * 400) return;
  int ox = i % 20, oy = (i / 20) % 20, oc = (i / 400) % 32, bb = i / 12800;
  float s = b[oc];
  for (int ic = 0; ic < 30; ++ic) {
    const float* ip = in + ((size_t)(bb * 30 + ic) * 84 + oy * 4) * 84 + ox * 4;
    const float* wp = w + (size_t)(oc * 30 + ic) * 64;
    for (int ky = 0; ky < 8; ++ky)
#pragma unroll
      for (int kx = 0; kx < 8; ++kx) s += ip[ky * 84 + kx] * wp[ky * 8 + kx];
  }
  out[i] = fmaxf(s, 0.0f);
}
__global__ __launch_bounds__(256)
void conv2_k(const float* __restrict__ in, const float* __restrict__ w,
             const float* __restrict__ b, float* __restrict__ out)
{
  int i = blockIdx.x * 256 + threadIdx.x;        // B*64*9*9
  if (i >= B_ * 64 * 81) return;
  int ox = i % 9, oy = (i / 9) % 9, oc = (i / 81) % 64, bb = i / 5184;
  float s = b[oc];
  for (int ic = 0; ic < 32; ++ic) {
    const float* ip = in + ((size_t)(bb * 32 + ic) * 20 + oy * 2) * 20 + ox * 2;
    const float* wp = w + (size_t)(oc * 32 + ic) * 16;
#pragma unroll
    for (int ky = 0; ky < 4; ++ky)
#pragma unroll
      for (int kx = 0; kx < 4; ++kx) s += ip[ky * 20 + kx] * wp[ky * 4 + kx];
  }
  out[i] = fmaxf(s, 0.0f);
}
__global__ __launch_bounds__(256)
void conv3_k(const float* __restrict__ in, const float* __restrict__ w,
             const float* __restrict__ b, uint16_t* __restrict__ flat)
{
  int i = blockIdx.x * 256 + threadIdx.x;        // B*64*7*7
  if (i >= B_ * 64 * 49) return;
  int ox = i % 7, oy = (i / 7) % 7, oc = (i / 49) % 64, bb = i / FLAT_;
  float s = b[oc];
  for (int ic = 0; ic < 64; ++ic) {
    const float* ip = in + ((size_t)(bb * 64 + ic) * 9 + oy) * 9 + ox;
    const float* wp = w + (size_t)(oc * 64 + ic) * 9;
#pragma unroll
    for (int ky = 0; ky < 3; ++ky)
#pragma unroll
      for (int kx = 0; kx < 3; ++kx) s += ip[ky * 9 + kx] * wp[ky * 3 + kx];
  }
  flat[(size_t)bb * FLAT_ + oc * 49 + oy * 7 + ox] = f2bfu(fmaxf(s, 0.0f));
}

// ============================================================
// Token assembly (r/s/a + time emb, modality-major) fused with embedding LN
// One block per (b, row) of x (B, 3T, H) f32.
// ============================================================
__global__ __launch_bounds__(256)
void assemble_ln(const int* __restrict__ timesteps, const int* __restrict__ actions,
                 const float* __restrict__ rtg, const float* __restrict__ rtg_w,
                 const float* __restrict__ rtg_b, const float* __restrict__ emb_t,
                 const float* __restrict__ emb_a, const float* __restrict__ s_emb,
                 const float* __restrict__ eg, const float* __restrict__ eb,
                 float* __restrict__ x)
{
  __shared__ float red[256];
  __shared__ float rrow[T_];
  const int bid = blockIdx.x;                    // 0..ROWS_-1
  const int b = bid / T3_, r = bid % T3_;
  const int m = r / T_, t = r % T_;
  const int tid = threadIdx.x;
  if (tid < T_) rrow[tid] = rtg[b * T_ + tid];
  __syncthreads();
  const int ts = timesteps[b * T_ + t];
  const int ac = actions[b * T_ + t];
  float v[3];
#pragma unroll
  for (int j = 0; j < 3; ++j) {
    int hh = tid + j * 256;
    float te = tanhf(emb_t[(size_t)ts * H_ + hh]);
    float e;
    if (m == 0) {
      float s = rtg_b[t * H_ + hh];
      for (int k = 0; k < T_; ++k) s += rrow[k] * rtg_w[(size_t)k * HT_ + t * H_ + hh];
      e = tanhf(s);
    } else if (m == 1) {
      e = s_emb[(size_t)b * HT_ + t * H_ + hh];  // already tanh'd in GEMM epilogue
    } else {
      e = tanhf(emb_a[ac * H_ + hh]);
    }
    v[j] = e + te;
  }
  red[tid] = v[0] + v[1] + v[2];
  __syncthreads();
  for (int o = 128; o; o >>= 1) { if (tid < o) red[tid] += red[tid + o]; __syncthreads(); }
  float mu = red[0] * (1.0f / 768.0f);
  __syncthreads();
  float d0 = v[0] - mu, d1 = v[1] - mu, d2 = v[2] - mu;
  red[tid] = d0 * d0 + d1 * d1 + d2 * d2;
  __syncthreads();
  for (int o = 128; o; o >>= 1) { if (tid < o) red[tid] += red[tid + o]; __syncthreads(); }
  float rs = rsqrtf(red[0] * (1.0f / 768.0f) + 1e-5f);
  float* xr = x + (size_t)bid * H_;
  xr[tid]       = eg[tid]       * d0 * rs + eb[tid];
  xr[tid + 256] = eg[tid + 256] * d1 * rs + eb[tid + 256];
  xr[tid + 512] = eg[tid + 512] * d2 * rs + eb[tid + 512];
}

// Generic LayerNorm over H=768, f32 in -> bf16 out (one block per row)
__global__ __launch_bounds__(256)
void ln_rows(const float* __restrict__ x, const float* __restrict__ g,
             const float* __restrict__ bta, uint16_t* __restrict__ out)
{
  __shared__ float red[256];
  const int row = blockIdx.x, tid = threadIdx.x;
  const float* xr = x + (size_t)row * H_;
  float v0 = xr[tid], v1 = xr[tid + 256], v2 = xr[tid + 512];
  red[tid] = v0 + v1 + v2;
  __syncthreads();
  for (int o = 128; o; o >>= 1) { if (tid < o) red[tid] += red[tid + o]; __syncthreads(); }
  float mu = red[0] * (1.0f / 768.0f);
  __syncthreads();
  float d0 = v0 - mu, d1 = v1 - mu, d2 = v2 - mu;
  red[tid] = d0 * d0 + d1 * d1 + d2 * d2;
  __syncthreads();
  for (int o = 128; o; o >>= 1) { if (tid < o) red[tid] += red[tid + o]; __syncthreads(); }
  float rs = rsqrtf(red[0] * (1.0f / 768.0f) + 1e-5f);
  uint16_t* orow = out + (size_t)row * H_;
  orow[tid]       = f2bfu(g[tid]       * d0 * rs + bta[tid]);
  orow[tid + 256] = f2bfu(g[tid + 256] * d1 * rs + bta[tid + 256]);
  orow[tid + 512] = f2bfu(g[tid + 512] * d2 * rs + bta[tid + 512]);
}

// gather channel-2 rows (r = 3t+2) into compact (B*T, H) bf16 for state head
__global__ __launch_bounds__(256)
void gather_ch2(const uint16_t* __restrict__ hf, uint16_t* __restrict__ out)
{
  int idx = blockIdx.x * 256 + threadIdx.x;      // B*T*H
  if (idx >= B_ * T_ * H_) return;
  int i = idx / H_, hh = idx % H_;
  int b = i / T_, t = i % T_;
  out[idx] = hf[((size_t)(b * T3_ + 3 * t + 2)) * H_ + hh];
}

// tiny action / return heads (VALU)
__global__ __launch_bounds__(32)
void small_heads(const uint16_t* __restrict__ hf, const float* __restrict__ aw,
                 const float* __restrict__ ab, const float* __restrict__ rw,
                 const float* __restrict__ rb, float* __restrict__ out_act,
                 float* __restrict__ out_rtg)
{
  int i = blockIdx.x;                            // B*T
  int b = i / T_, t = i % T_;
  int tid = threadIdx.x;
  if (tid < 4) {
    const uint16_t* row = hf + (size_t)(b * T3_ + 3 * t + 1) * H_;
    float s = ab[tid];
    for (int hh = 0; hh < H_; ++hh) s += bfu2f(row[hh]) * aw[hh * 4 + tid];
    out_act[i * 4 + tid] = s;
  } else if (tid == 4) {
    const uint16_t* row = hf + (size_t)(b * T3_ + 3 * t + 2) * H_;
    float s = rb[0];
    for (int hh = 0; hh < H_; ++hh) s += bfu2f(row[hh]) * rw[hh];
    out_rtg[i] = s;
  }
}

// ============================================================
// host-side orchestration
// ============================================================
enum {
  I_TIMESTEPS = 0, I_STATES, I_ACTIONS, I_RTG,
  I_EMB_T, I_RTG_W, I_RTG_B,
  I_CONV1_W, I_CONV1_B, I_CONV2_W, I_CONV2_B, I_CONV3_W, I_CONV3_B,
  I_ST_W, I_ST_B, I_EMB_A, I_ELN_G, I_ELN_B,
  I_LN1_G, I_LN1_B, I_WQKV, I_BQKV, I_WPROJ, I_BPROJ,
  I_LN2_G, I_LN2_B, I_WFC, I_BFC, I_WFC2, I_BFC2,
  I_LNF_G, I_LNF_B, I_RTG_HW, I_RTG_HB, I_ST_HW, I_ST_HB, I_ACT_HW, I_ACT_HB
};

extern "C" void kernel_launch(void* const* d_in, const int* in_sizes, int n_in,
                              void* d_out, int out_size, void* d_ws, size_t ws_size,
                              hipStream_t stream)
{
  (void)in_sizes; (void)n_in; (void)out_size; (void)ws_size;
  const int*   timesteps = (const int*)d_in[I_TIMESTEPS];
  const float* states    = (const float*)d_in[I_STATES];
  const int*   actions   = (const int*)d_in[I_ACTIONS];
  const float* rtg       = (const float*)d_in[I_RTG];
  const float* emb_t     = (const float*)d_in[I_EMB_T];
  const float* rtg_w     = (const float*)d_in[I_RTG_W];
  const float* rtg_b     = (const float*)d_in[I_RTG_B];
  const float* c1w = (const float*)d_in[I_CONV1_W], *c1b = (const float*)d_in[I_CONV1_B];
  const float* c2w = (const float*)d_in[I_CONV2_W], *c2b = (const float*)d_in[I_CONV2_B];
  const float* c3w = (const float*)d_in[I_CONV3_W], *c3b = (const float*)d_in[I_CONV3_B];
  const float* st_w = (const float*)d_in[I_ST_W],   *st_b = (const float*)d_in[I_ST_B];
  const float* emb_a = (const float*)d_in[I_EMB_A];
  const float* eln_g = (const float*)d_in[I_ELN_G], *eln_b = (const float*)d_in[I_ELN_B];
  const float* ln1_g = (const float*)d_in[I_LN1_G], *ln1_b = (const float*)d_in[I_LN1_B];
  const float* wqkv  = (const float*)d_in[I_WQKV],  *bqkv  = (const float*)d_in[I_BQKV];
  const float* wproj = (const float*)d_in[I_WPROJ], *bproj = (const float*)d_in[I_BPROJ];
  const float* ln2_g = (const float*)d_in[I_LN2_G], *ln2_b = (const float*)d_in[I_LN2_B];
  const float* wfc   = (const float*)d_in[I_WFC],   *bfc   = (const float*)d_in[I_BFC];
  const float* wfc2  = (const float*)d_in[I_WFC2],  *bfc2  = (const float*)d_in[I_BFC2];
  const float* lnf_g = (const float*)d_in[I_LNF_G], *lnf_b = (const float*)d_in[I_LNF_B];
  const float* rtg_hw = (const float*)d_in[I_RTG_HW], *rtg_hb = (const float*)d_in[I_RTG_HB];
  const float* st_hw  = (const float*)d_in[I_ST_HW],  *st_hb  = (const float*)d_in[I_ST_HB];
  const float* act_hw = (const float*)d_in[I_ACT_HW], *act_hb = (const float*)d_in[I_ACT_HB];

  // ---- workspace layout (bytes) ----
  char* ws = (char*)d_ws;
  const size_t off_x    = 0;                                  // 11520*768 f32
  const size_t off_aB   = off_x  + (size_t)ROWS_ * H_ * 4;    // LN out bf16 / conv scratch
  const size_t off_qkv  = off_aB + 22000000;                  // 11520*2304 bf16
  const size_t off_y    = off_qkv + (size_t)ROWS_ * H3_ * 2;  // 11520*768 bf16
  const size_t off_hf2  = off_y  + (size_t)ROWS_ * H_ * 2;    // 3840*768 bf16
  float*    xb   = (float*)(ws + off_x);
  uint16_t* aB   = (uint16_t*)(ws + off_aB);
  uint16_t* qkvB = (uint16_t*)(ws + off_qkv);
  uint16_t* yB   = (uint16_t*)(ws + off_y);
  uint16_t* ffB  = (uint16_t*)(ws + off_qkv);   // alias qkv+y (dead when used)
  uint16_t* hf2  = (uint16_t*)(ws + off_hf2);
  // conv scratch aliases inside aB region (all pre-transformer)
  float*    h1    = (float*)(ws + off_aB);                       // 6,553,600 B
  float*    h2    = (float*)(ws + off_aB + 6553600);             // 2,654,208 B
  uint16_t* flatA = (uint16_t*)(ws + off_aB + 6553600 + 2654208);// 802,816 B
  float*    s_emb = (float*)(ws + off_aB + 6553600 + 2654208 + 802816); // 11,796,480 B

  float* out_state = (float*)d_out;                              // B*T*84*84
  float* out_act   = out_state + (size_t)B_ * T_ * SD2_;         // B*T*4
  float* out_rtg   = out_act + (size_t)B_ * T_ * 4;              // B*T

  // ---- conv encoder ----
  conv1_k<<<(B_ * 32 * 400 + 255) / 256, 256, 0, stream>>>(states, c1w, c1b, h1);
  conv2_k<<<(B_ * 64 * 81 + 255) / 256, 256, 0, stream>>>(h1, c2w, c2b, h2);
  conv3_k<<<(B_ * 64 * 49 + 255) / 256, 256, 0, stream>>>(h2, c3w, c3b, flatA);

  // state embedding GEMM: (128 x 3136) @ (3136 x 23040) + b, tanh -> s_emb f32
  gemm_wmma<<<dim3(HT_ / 128, 1), 256, 0, stream>>>(
      flatA, st_w, st_b, nullptr, s_emb, nullptr, B_, HT_, FLAT_, 1);

  // token assembly + embedding LN -> x (B, 3T, H) f32
  assemble_ln<<<ROWS_, 256, 0, stream>>>(timesteps, actions, rtg, rtg_w, rtg_b,
                                         emb_t, emb_a, s_emb, eln_g, eln_b, xb);

  // ---- transformer ----
  for (int l = 0; l < L_; ++l) {
    const float* wq = wqkv  + (size_t)l * H_ * H3_;
    const float* bq = bqkv  + (size_t)l * H3_;
    const float* wp = wproj + (size_t)l * H_ * H_;
    const float* bp = bproj + (size_t)l * H_;
    const float* wf = wfc   + (size_t)l * H_ * H4_;
    const float* bf = bfc   + (size_t)l * H4_;
    const float* w2 = wfc2  + (size_t)l * H4_ * H_;
    const float* b2 = bfc2  + (size_t)l * H_;

    ln_rows<<<ROWS_, 256, 0, stream>>>(xb, ln1_g + l * H_, ln1_b + l * H_, aB);
    gemm_wmma<<<dim3(H3_ / 128, ROWS_ / 128), 256, 0, stream>>>(
        aB, wq, bq, nullptr, nullptr, qkvB, ROWS_, H3_, H_, 0);
    attn_wmma<<<B_ * NH_, 256, 0, stream>>>(qkvB, yB);
    gemm_wmma<<<dim3(H_ / 128, ROWS_ / 128), 256, 0, stream>>>(
        yB, wp, bp, xb, xb, nullptr, ROWS_, H_, H_, 0);
    ln_rows<<<ROWS_, 256, 0, stream>>>(xb, ln2_g + l * H_, ln2_b + l * H_, aB);
    gemm_wmma<<<dim3(H4_ / 128, ROWS_ / 128), 256, 0, stream>>>(
        aB, wf, bf, nullptr, nullptr, ffB, ROWS_, H4_, H_, 2);
    gemm_wmma<<<dim3(H_ / 128, ROWS_ / 128), 256, 0, stream>>>(
        ffB, w2, b2, xb, xb, nullptr, ROWS_, H_, H4_, 0);
  }

  // ---- final LN + heads ----
  ln_rows<<<ROWS_, 256, 0, stream>>>(xb, lnf_g, lnf_b, aB);
  gather_ch2<<<(B_ * T_ * H_ + 255) / 256, 256, 0, stream>>>(aB, hf2);
  gemm_wmma<<<dim3((SD2_ + 127) / 128, (B_ * T_) / 128), 256, 0, stream>>>(
      hf2, st_hw, st_hb, nullptr, out_state, nullptr, B_ * T_, SD2_, H_, 0);
  small_heads<<<B_ * T_, 32, 0, stream>>>(aB, act_hw, act_hb, rtg_hw, rtg_hb,
                                          out_act, out_rtg);
}
